// SeparableATT4_17016660427252
// MI455X (gfx1250) — matrix operand order repaired and checked by reading, not compiled, not run on hardware
//
#include <hip/hip_runtime.h>

// ---------------------------------------------------------------------------
// SeparableATT4 for gfx1250 (CDNA5, wave32, WMMA 16x16x32 f16 -> f32)
//  - fused projection GEMM (double-buffered LDS, async global->LDS for A)
//  - split-K energy GEMM (deterministic, reduction fused into softmax)
// ---------------------------------------------------------------------------

#define CIN   512
#define NTOT  4096
#define MTOT  1152   // 256(Q) + 256(K) + 512(V) + 64(Qs) + 64(Ke)
#define NBAT  16
#define KSPLIT 8     // split-K factor for energy GEMM

typedef _Float16 half16 __attribute__((ext_vector_type(16)));
typedef _Float16 half8  __attribute__((ext_vector_type(8)));
typedef _Float16 half4  __attribute__((ext_vector_type(4)));
typedef float    float8 __attribute__((ext_vector_type(8)));

union FragAB { half16 f; half8 h[2]; };

__device__ __forceinline__ float8 wmma_f16(half16 a, half16 b, float8 c) {
  // D = A(16x32 f16) * B(32x16 f16) + C(16x16 f32)
  return __builtin_amdgcn_wmma_f32_16x16x32_f16(
      /*neg_a=*/false, a, /*neg_b=*/false, b,
      /*c_mod=*/(short)0, c, /*reuse_a=*/false, /*reuse_b=*/false);
}

// CDNA5 async copy: global -> LDS, 16B per lane, tracked by ASYNCcnt.
__device__ __forceinline__ void async_load_b128(unsigned lds_off, const void* gaddr) {
  asm volatile("global_load_async_to_lds_b128 %0, %1, off"
               :: "v"(lds_off), "v"((unsigned long long)(uintptr_t)gaddr)
               : "memory");
}
__device__ __forceinline__ void wait_asynccnt0() {
  asm volatile("s_wait_asynccnt 0" ::: "memory");
}

// ---------------------------------------------------------------------------
// Kernel 1: pack [Wq;Wk;Wv;Ws;We] -> f16 W_all[1152][512], biases -> f32[1152]
// ---------------------------------------------------------------------------
__global__ __launch_bounds__(256) void pack_weights(
    const float* __restrict__ Wq, const float* __restrict__ bq,
    const float* __restrict__ Wk, const float* __restrict__ bk,
    const float* __restrict__ Wv, const float* __restrict__ bv,
    const float* __restrict__ Ws, const float* __restrict__ bs,
    const float* __restrict__ We, const float* __restrict__ be,
    _Float16* __restrict__ Wh, float* __restrict__ bias) {
  int idx = blockIdx.x * 256 + threadIdx.x;
  if (idx >= MTOT * CIN) return;
  int r = idx / CIN, c = idx - r * CIN;
  const float* W; const float* bb; int rr;
  if      (r < 256)  { W = Wq; bb = bq; rr = r;        }
  else if (r < 512)  { W = Wk; bb = bk; rr = r - 256;  }
  else if (r < 1024) { W = Wv; bb = bv; rr = r - 512;  }
  else if (r < 1088) { W = Ws; bb = bs; rr = r - 1024; }
  else               { W = We; bb = be; rr = r - 1088; }
  Wh[idx] = (_Float16)W[(size_t)rr * CIN + c];
  if (c == 0) bias[r] = bb[rr];
}

// ---------------------------------------------------------------------------
// Kernel 2: P[b] = W_all * x[b] + bias   (M=1152, K=512, N=4096), P in f16
//   block = 256 threads (8 waves), tile 128x128, K-step 32, double-buffered.
//   A tile staged via async global->LDS (f16 already); B tile cvt f32->f16.
//   wave w: 32(M) x 64(N) = 2x4 WMMA tiles
// ---------------------------------------------------------------------------
__global__ __launch_bounds__(256) void gemm_proj(
    const _Float16* __restrict__ Wh, const float* __restrict__ bias,
    const float* __restrict__ x, _Float16* __restrict__ P) {
  const int b     = blockIdx.z;
  const int mbase = blockIdx.y * 128;
  const int nbase = blockIdx.x * 128;

  __shared__ _Float16 Als[2][128][40];   // 128 x 32, padded, double buffer
  __shared__ _Float16 Bls[2][32][136];   // 32 x 128, padded, double buffer

  const int tid  = threadIdx.x;
  const int lane = tid & 31;
  const int wv   = tid >> 5;
  const int wm0  = (wv & 3) * 32;     // wave M origin in tile
  const int wn0  = (wv >> 2) * 64;    // wave N origin in tile

  float8 acc[2][4] = {};
  const float* xb = x + (size_t)b * CIN * NTOT;

  auto stageA = [&](int buf, int kb) {
#pragma unroll
    for (int i = 0; i < 2; ++i) {
      int chunk = tid + i * 256;            // 0..511
      int m = chunk >> 2, kc = (chunk & 3) * 8;
      async_load_b128((unsigned)(uintptr_t)&Als[buf][m][kc],
                      Wh + (size_t)(mbase + m) * CIN + kb + kc);
    }
  };
  auto stageB = [&](int buf, int kb) {
#pragma unroll
    for (int i = 0; i < 4; ++i) {
      int chunk = tid + i * 256;            // 0..1023
      int r = chunk >> 5, cc = (chunk & 31) * 4;
      float4 v = *(const float4*)(xb + (size_t)(kb + r) * NTOT + nbase + cc);
      half4 hv;
      hv[0] = (_Float16)v.x; hv[1] = (_Float16)v.y;
      hv[2] = (_Float16)v.z; hv[3] = (_Float16)v.w;
      *(half4*)&Bls[buf][r][cc] = hv;
    }
  };

  stageA(0, 0);
  stageB(0, 0);
  wait_asynccnt0();
  __syncthreads();

  for (int step = 0; step < CIN / 32; ++step) {
    const int cur = step & 1;
    if (step + 1 < CIN / 32) {            // prefetch next K-step into other buffer
      stageA(1 - cur, (step + 1) * 32);
      stageB(1 - cur, (step + 1) * 32);
    }

    // --- fragments + WMMA from current buffer ---
    FragAB a[2];
    const int klo = (lane >> 4) * 8;
#pragma unroll
    for (int i = 0; i < 2; ++i) {
      int mrow = wm0 + i * 16 + (lane & 15);
      a[i].h[0] = *(const half8*)&Als[cur][mrow][klo];
      a[i].h[1] = *(const half8*)&Als[cur][mrow][klo + 16];
    }
#pragma unroll
    for (int j = 0; j < 4; ++j) {
      FragAB bf;
      int ncol = wn0 + j * 16;
#pragma unroll
      for (int s = 0; s < 16; ++s)
        bf.f[s] = Bls[cur][klo + (s & 7) + ((s >> 3) << 4)][ncol + (lane & 15)];
#pragma unroll
      for (int i = 0; i < 2; ++i) acc[i][j] = wmma_f16(a[i].f, bf.f, acc[i][j]);
    }

    wait_asynccnt0();    // our async A-stores to the other buffer are in LDS
    __syncthreads();     // everyone done staging next + reading current
  }

  // --- epilogue: + bias, convert f16, store P ---
  _Float16* Pb = P + (size_t)b * MTOT * NTOT;
#pragma unroll
  for (int i = 0; i < 2; ++i)
#pragma unroll
    for (int j = 0; j < 4; ++j)
#pragma unroll
      for (int r = 0; r < 8; ++r) {
        int m = mbase + wm0 + i * 16 + r + ((lane >> 4) << 3);
        int n = nbase + wn0 + j * 16 + (lane & 15);
        Pb[(size_t)m * NTOT + n] = (_Float16)(acc[i][j][r] + bias[m]);
      }
}

// ---------------------------------------------------------------------------
// Kernel 3: gate = sigmoid(mean_c Q*K); V *= gate (in place, rows 512..1023)
// ---------------------------------------------------------------------------
__global__ __launch_bounds__(256) void gate_kernel(_Float16* __restrict__ P) {
  const int b = blockIdx.x >> 4;
  const int n = (blockIdx.x & 15) * 256 + threadIdx.x;
  _Float16* Pb = P + (size_t)b * MTOT * NTOT;
  float s = 0.f;
  for (int c = 0; c < 256; ++c) {
    float q = (float)Pb[(size_t)c * NTOT + n];
    float k = (float)Pb[(size_t)(256 + c) * NTOT + n];
    s += q * k;
  }
  float g = 1.f / (1.f + __expf(-s * (1.0f / 256.0f)));
  for (int c = 0; c < 512; ++c) {
    size_t idx = (size_t)(512 + c) * NTOT + n;
    Pb[idx] = (_Float16)((float)Pb[idx] * g);
  }
}

// ---------------------------------------------------------------------------
// Kernel 4: energy partials: Ep[s][b] = V_dw * Qs^T over K-chunk s
//   grid (4 Mtiles, 16 batches, 8 K-splits) -> 512 blocks
//   wave w -> M-tile w (16 rows) x 4 N-tiles; K chunk = 512
// ---------------------------------------------------------------------------
__global__ __launch_bounds__(256) void energy_gemm(
    const _Float16* __restrict__ P, float* __restrict__ Epart) {
  const int b     = blockIdx.y;
  const int mbase = blockIdx.x * 128;
  const int spl   = blockIdx.z;
  const int nk0   = spl * (NTOT / KSPLIT);
  const _Float16* Pv = P + (size_t)b * MTOT * NTOT + (size_t)512  * NTOT;
  const _Float16* Pq = P + (size_t)b * MTOT * NTOT + (size_t)1024 * NTOT;

  __shared__ _Float16 Bq[32][72];   // Qs^T tile: [k(n)][d], padded

  const int tid  = threadIdx.x;
  const int lane = tid & 31;
  const int wv   = tid >> 5;
  const int dcol = tid >> 2;          // 0..63
  const int jb   = (tid & 3) * 8;     // 0,8,16,24

  float8 acc[4] = {};

  for (int nk = nk0; nk < nk0 + NTOT / KSPLIT; nk += 32) {
    // transpose-stage Qs[64][nk..nk+32) -> Bq[k][d]
    half8 v = *(const half8*)(Pq + (size_t)dcol * NTOT + nk + jb);
#pragma unroll
    for (int e = 0; e < 8; ++e) Bq[jb + e][dcol] = v[e];
    __syncthreads();

    FragAB a;
    const int mrow = mbase + wv * 16 + (lane & 15);
    const int klo  = (lane >> 4) * 8;
    a.h[0] = *(const half8*)(Pv + (size_t)mrow * NTOT + nk + klo);
    a.h[1] = *(const half8*)(Pv + (size_t)mrow * NTOT + nk + klo + 16);
#pragma unroll
    for (int j = 0; j < 4; ++j) {
      FragAB bf;
#pragma unroll
      for (int s = 0; s < 16; ++s)
        bf.f[s] = Bq[klo + (s & 7) + ((s >> 3) << 4)][j * 16 + (lane & 15)];
      acc[j] = wmma_f16(a.f, bf.f, acc[j]);
    }
    __syncthreads();
  }

  // Epart layout: [s][b][512][64]
  float* Eb = Epart + ((size_t)spl * NBAT + b) * 512 * 64;
#pragma unroll
  for (int j = 0; j < 4; ++j)
#pragma unroll
    for (int r = 0; r < 8; ++r) {
      int m = mbase + wv * 16 + r + ((lane >> 4) << 3);
      int d = j * 16 + (lane & 15);
      Eb[(size_t)m * 64 + d] = acc[j][r];
    }
}

// ---------------------------------------------------------------------------
// Kernel 5: reduce split-K partials (fixed order -> deterministic) + softmax
//           over d=64; attention stored as f16
// ---------------------------------------------------------------------------
__global__ __launch_bounds__(256) void softmax_kernel(
    const float* __restrict__ Epart, _Float16* __restrict__ attn) {
  const int row  = blockIdx.x * 8 + (threadIdx.x >> 5);   // 0..8191 = b*512+c
  const int lane = threadIdx.x & 31;
  float v0 = 0.f, v1 = 0.f;
#pragma unroll
  for (int s = 0; s < KSPLIT; ++s) {
    const float* e = Epart + ((size_t)s * NBAT * 512 + row) * 64;
    v0 += e[lane];
    v1 += e[lane + 32];
  }
  float m = fmaxf(v0, v1);
#pragma unroll
  for (int off = 16; off > 0; off >>= 1) m = fmaxf(m, __shfl_xor(m, off, 32));
  v0 = __expf(v0 - m);
  v1 = __expf(v1 - m);
  float s = v0 + v1;
#pragma unroll
  for (int off = 16; off > 0; off >>= 1) s += __shfl_xor(s, off, 32);
  float inv = 1.f / s;
  attn[(size_t)row * 64 + lane]      = (_Float16)(v0 * inv);
  attn[(size_t)row * 64 + lane + 32] = (_Float16)(v1 * inv);
}

// ---------------------------------------------------------------------------
// Kernel 6: out[b] = x[b] + attention(512 x 64) * Ke(64 x 4096)
//   K=64 -> 2 WMMA K-steps; operands fragment-loaded directly from global
//   wave w -> M-tile w (16 rows) x 8 N-tiles (128 cols)
// ---------------------------------------------------------------------------
__global__ __launch_bounds__(256) void out_gemm(
    const _Float16* __restrict__ attn, const _Float16* __restrict__ P,
    const float* __restrict__ x, float* __restrict__ out) {
  const int b     = blockIdx.z;
  const int mbase = blockIdx.y * 128;
  const int nbase = blockIdx.x * 128;
  const int lane  = threadIdx.x & 31;
  const int wv    = threadIdx.x >> 5;

  const _Float16* Ab = attn + (size_t)b * 512 * 64;
  const _Float16* Ke = P + (size_t)b * MTOT * NTOT + (size_t)1088 * NTOT;

  float8 acc[8] = {};
#pragma unroll
  for (int ks = 0; ks < 2; ++ks) {
    FragAB a;
    const int mrow = mbase + wv * 16 + (lane & 15);
    const int klo  = ks * 32 + (lane >> 4) * 8;
    a.h[0] = *(const half8*)(Ab + (size_t)mrow * 64 + klo);
    a.h[1] = *(const half8*)(Ab + (size_t)mrow * 64 + klo + 16);
    const int krow = ks * 32 + lane;
#pragma unroll
    for (int j = 0; j < 8; ++j) {
      FragAB bf;
      int nc = nbase + j * 16;
#pragma unroll
      for (int s = 0; s < 16; ++s)
        bf.f[s] = Ke[(size_t)(klo + (s & 7) + ((s >> 3) << 4)) * NTOT + nc + (lane & 15)];
      acc[j] = wmma_f16(a.f, bf.f, acc[j]);
    }
  }

  const float* xb = x   + (size_t)b * CIN * NTOT;
  float*       ob = out + (size_t)b * CIN * NTOT;
#pragma unroll
  for (int j = 0; j < 8; ++j)
#pragma unroll
    for (int r = 0; r < 8; ++r) {
      int m = mbase + wv * 16 + r + ((lane >> 4) << 3);
      int n = nbase + j * 16 + (lane & 15);
      size_t idx = (size_t)m * NTOT + n;
      ob[idx] = xb[idx] + acc[j][r];
    }
}

// ---------------------------------------------------------------------------
// Host-side launcher
// ---------------------------------------------------------------------------
extern "C" void kernel_launch(void* const* d_in, const int* in_sizes, int n_in,
                              void* d_out, int out_size, void* d_ws, size_t ws_size,
                              hipStream_t stream) {
  (void)in_sizes; (void)n_in; (void)out_size; (void)ws_size;

  const float* x  = (const float*)d_in[0];
  const float* Wq = (const float*)d_in[1];
  const float* bq = (const float*)d_in[2];
  const float* Wk = (const float*)d_in[3];
  const float* bk = (const float*)d_in[4];
  const float* Wv = (const float*)d_in[5];
  const float* bv = (const float*)d_in[6];
  const float* Ws = (const float*)d_in[7];
  const float* bs = (const float*)d_in[8];
  const float* We = (const float*)d_in[9];
  const float* be = (const float*)d_in[10];

  // Workspace layout (256B aligned regions)
  char* ws = (char*)d_ws;
  _Float16* Wh   = (_Float16*)(ws + 0);              //   1,179,648 B
  float*    bias = (float*)   (ws + 1179648);        //       4,608 B
  _Float16* P    = (_Float16*)(ws + 1184256);        // 150,994,944 B (f16 [16][1152][4096])
  float*    Ep   = (float*)   (ws + 152179200);      //  16,777,216 B (f32 [8][16][512][64])
  _Float16* attn = (_Float16*)(ws + 168956416);      //   1,048,576 B (f16 [16][512][64])

  pack_weights<<<(MTOT * CIN + 255) / 256, 256, 0, stream>>>(
      Wq, bq, Wk, bk, Wv, bv, Ws, bs, We, be, Wh, bias);

  dim3 g1(NTOT / 128, MTOT / 128, NBAT);             // 32 x 9 x 16
  gemm_proj<<<g1, 256, 0, stream>>>(Wh, bias, x, P);

  gate_kernel<<<NBAT * (NTOT / 256), 256, 0, stream>>>(P);

  dim3 g3(512 / 128, NBAT, KSPLIT);                  // 4 x 16 x 8
  energy_gemm<<<g3, 256, 0, stream>>>(P, Ep);

  softmax_kernel<<<NBAT * 512 / 8, 256, 0, stream>>>(Ep, attn);

  dim3 g5(NTOT / 128, 512 / 128, NBAT);              // 32 x 4 x 16
  out_gemm<<<g5, 256, 0, stream>>>(attn, P, x, (float*)d_out);
}
